// LinkPredictionModel_37383395345042
// MI455X (gfx1250) — compile-verified, matrix operand order checked by reading, not run on hardware
//
#include <hip/hip_runtime.h>

// ---------------------------------------------------------------------------
// GCN (2x GCNConv + linear) for MI455X / gfx1250, wave32 + WMMA bf16.
//   GEMMs: v_wmma_f32_16x16x32_bf16, 128x128 block tile, BK=32, 8 waves,
//   double-buffered LDS, software-pipelined: stage(kt+1) overlaps compute(kt),
//   ONE barrier per tile. K-tail peeled OUT of the hot loop (guarded, once).
//   Aggregation: edge-parallel fp32 atomics (bandwidth bound, ~680MB/layer).
// ---------------------------------------------------------------------------

typedef __attribute__((ext_vector_type(16))) __bf16 v16bf;
typedef __attribute__((ext_vector_type(8)))  __bf16 v8bf;
typedef __attribute__((ext_vector_type(8)))  float  v8f;

#define N_NODES 10000
#define N_EDGES 320000
#define DIM     256

#define BM  128
#define BN  128
#define BK  32
#define LDT 40   // padded LDS K-stride (bf16 elems): 80B rows -> 16B-aligned frags

__device__ __forceinline__ unsigned bf16bits(float f) {
  union { float f; unsigned u; } x; x.f = f;
  return (x.u + 0x7FFFu + ((x.u >> 16) & 1u)) >> 16;      // RNE, low 16 bits
}
__device__ __forceinline__ unsigned pack2(float a, float b) {
  union { float f; unsigned u; } x, y; x.f = a; y.f = b;
  unsigned ua = (x.u + 0x7FFFu + ((x.u >> 16) & 1u)) >> 16;
  unsigned ub = (y.u + 0x7FFFu + ((y.u >> 16) & 1u)) & 0xFFFF0000u;
  return (ua & 0xFFFFu) | ub;
}
__device__ __forceinline__ __bf16 f2bf(float f) {
  union { unsigned short s; __bf16 b; } y; y.s = (unsigned short)bf16bits(f);
  return y.b;
}

// C = A[M,K] @ B[K,DIM] (row-major), optional +bias / ReLU epilogue.
__global__ __launch_bounds__(256)
void gemm_bf16_wmma(const float* __restrict__ A, int lda,
                    const float* __restrict__ B, int ldb,
                    float* __restrict__ C, int ldc,
                    int M, int K,
                    const float* __restrict__ bias, int relu)
{
  __shared__ __attribute__((aligned(16))) __bf16 As[2][BM * LDT];
  __shared__ __attribute__((aligned(16))) __bf16 Bs[2][BN * LDT];

  const int tid  = threadIdx.x;
  const int wid  = tid >> 5;
  const int lane = tid & 31;
  const int wm   = wid >> 1;          // 0..3  (32 rows each)
  const int wn   = wid & 1;           // 0..1  (64 cols each)
  const int rowBlock = blockIdx.x * BM;
  const int colBlock = blockIdx.y * BN;

  v8f acc[2][4];
  #pragma unroll
  for (int mi = 0; mi < 2; ++mi)
    #pragma unroll
    for (int ni = 0; ni < 4; ++ni)
      #pragma unroll
      for (int j = 0; j < 8; ++j) acc[mi][ni][j] = 0.0f;

  const int laneRow = lane & 15;
  const int kbase   = (lane >> 4) * 8;   // 16-bit frag: K 0-7/16-23 vs 8-15/24-31

  // A staging geometry (row-clamped, never OOB; clamped rows are never stored)
  const int aCol  = (tid & 7) * 4;                 // 0,4,...,28
  const int aRow0 = tid >> 3;                      // 0..31, +32 per step
  int aR[4];
  #pragma unroll
  for (int i = 0; i < 4; ++i) {
    int gr = rowBlock + aRow0 + i * 32;
    aR[i] = gr < M ? gr : (M - 1);
  }
  // B staging geometry: thread owns LDS row nn, 16 k-values
  const int bN    = tid & 127;
  const int bKoff = (tid >> 7) * 16;               // 0 or 16
  const int bCol  = colBlock + bN;                 // always < DIM (DIM % BN == 0)

  const int kFull = K / BK;             // >= 1 for all our GEMMs (K in {10000,256})
  const int kRem  = K - kFull * BK;     // tail (16 for K=10000, 0 for K=256)

  // ---- unguarded full-tile staging (hot path) ----
  auto stage_full = [&](int kt, int buf) {
    const int k0 = kt * BK;
    __bf16* as = As[buf];
    __bf16* bs = Bs[buf];
    // A tile: 4x float4 -> 4x 8B LDS stores
    #pragma unroll
    for (int i = 0; i < 4; ++i) {
      const float4 v = *(const float4*)(A + (size_t)aR[i] * lda + k0 + aCol);
      uint2 p; p.x = pack2(v.x, v.y); p.y = pack2(v.z, v.w);
      *(uint2*)&as[(aRow0 + i * 32) * LDT + aCol] = p;
    }
    // B tile (transposed to Bs[n][k]): 16 coalesced b32 loads -> 2x b128 stores
    const float* bp = B + (size_t)(k0 + bKoff) * ldb + bCol;
    float f[16];
    #pragma unroll
    for (int j = 0; j < 16; ++j) f[j] = bp[(size_t)j * ldb];
    unsigned u[8];
    #pragma unroll
    for (int j = 0; j < 8; ++j) u[j] = pack2(f[2 * j], f[2 * j + 1]);
    uint4* q = (uint4*)&bs[bN * LDT + bKoff];
    q[0] = uint4{u[0], u[1], u[2], u[3]};
    q[1] = uint4{u[4], u[5], u[6], u[7]};
  };

  // ---- guarded K-tail staging (outside hot loop, runs at most once) ----
  auto stage_tail = [&](int buf) {
    const int k0 = kFull * BK;
    __bf16* as = As[buf];
    __bf16* bs = Bs[buf];
    #pragma unroll
    for (int i = 0; i < 16; ++i) {
      int idx = tid + i * 256;
      int r = idx >> 5, c = idx & 31;
      int gr = rowBlock + r; gr = gr < M ? gr : (M - 1);
      float v = (c < kRem) ? A[(size_t)gr * lda + k0 + c] : 0.0f;
      as[r * LDT + c] = f2bf(v);
    }
    #pragma unroll
    for (int i = 0; i < 16; ++i) {
      int idx = tid + i * 256;
      int kk = idx >> 7, nn = idx & 127;
      float v = (kk < kRem) ? B[(size_t)(k0 + kk) * ldb + colBlock + nn] : 0.0f;
      bs[nn * LDT + kk] = f2bf(v);
    }
  };

  // ---- fragment load + 8x WMMA from buffer buf ----
  auto compute_step = [&](int buf) {
    const __bf16* as = As[buf];
    const __bf16* bs = Bs[buf];
    v16bf a[2], b[4];
    #pragma unroll
    for (int mi = 0; mi < 2; ++mi) {
      const __bf16* p = &as[(wm * 32 + mi * 16 + laneRow) * LDT + kbase];
      v8bf lo = *(const v8bf*)p;
      v8bf hi = *(const v8bf*)(p + 16);
      #pragma unroll
      for (int j = 0; j < 8; ++j) { a[mi][j] = lo[j]; a[mi][j + 8] = hi[j]; }
    }
    #pragma unroll
    for (int ni = 0; ni < 4; ++ni) {
      const __bf16* p = &bs[(wn * 64 + ni * 16 + laneRow) * LDT + kbase];
      v8bf lo = *(const v8bf*)p;
      v8bf hi = *(const v8bf*)(p + 16);
      #pragma unroll
      for (int j = 0; j < 8; ++j) { b[ni][j] = lo[j]; b[ni][j + 8] = hi[j]; }
    }
    #pragma unroll
    for (int mi = 0; mi < 2; ++mi)
      #pragma unroll
      for (int ni = 0; ni < 4; ++ni)
        acc[mi][ni] = __builtin_amdgcn_wmma_f32_16x16x32_bf16(
            false, a[mi], false, b[ni], (short)0, acc[mi][ni], false, false);
  };

  // ---- pipelined main loop over FULL tiles only (no tail code inside) ----
  stage_full(0, 0);
  for (int kt = 0; kt + 1 < kFull; ++kt) {
    __syncthreads();                       // buf[kt&1] staged; prior reads done
    stage_full(kt + 1, (kt + 1) & 1);
    compute_step(kt & 1);
  }
  // last full tile: overlap guarded tail staging (if any), then drain
  __syncthreads();
  if (kRem > 0) stage_tail(kFull & 1);
  compute_step((kFull - 1) & 1);
  if (kRem > 0) {
    __syncthreads();
    compute_step(kFull & 1);
  }

  // ================= epilogue =================
  const int rlane = (lane >> 4) * 8;
  const bool fullBlock = (rowBlock + BM <= M);
  #pragma unroll
  for (int mi = 0; mi < 2; ++mi) {
    #pragma unroll
    for (int ni = 0; ni < 4; ++ni) {
      int c = colBlock + wn * 64 + ni * 16 + laneRow;
      float bv = bias ? bias[c] : 0.0f;
      #pragma unroll
      for (int j = 0; j < 8; ++j) {
        int r = rowBlock + wm * 32 + mi * 16 + rlane + j;
        float v = acc[mi][ni][j] + bv;
        if (relu) v = v > 0.0f ? v : 0.0f;
        if (fullBlock || r < M) C[(size_t)r * ldc + c] = v;
      }
    }
  }
}

// ---------------- graph / elementwise kernels ----------------

__global__ void k_fill1(float* p, int n) {
  int i = blockIdx.x * blockDim.x + threadIdx.x;
  if (i < n) p[i] = 1.0f;                      // self-loop degree
}

__global__ void k_deg(const int* __restrict__ dst, float* __restrict__ deg, int e) {
  int i = blockIdx.x * blockDim.x + threadIdx.x;
  if (i < e) atomicAdd(&deg[dst[i]], 1.0f);
}

__global__ void k_rsqrt(float* p, int n) {
  int i = blockIdx.x * blockDim.x + threadIdx.x;
  if (i < n) p[i] = rsqrtf(p[i]);              // deg >= 1 always
}

// out[n,d] = h[n,d]*dinv[n]^2 + bias[d]   (self-loop term + bias init)
__global__ void k_self_bias(const float* __restrict__ h, const float* __restrict__ dinv,
                            const float* __restrict__ bias, float* __restrict__ out, int n) {
  int i = blockIdx.x * blockDim.x + threadIdx.x;
  if (i < n) {
    int node = i >> 8;
    float di = dinv[node];
    out[i] = h[i] * di * di + bias[i & (DIM - 1)];
  }
}

// one block (256 threads) per edge: out[dst] += h[src] * dinv[src]*dinv[dst]
__global__ __launch_bounds__(256)
void k_edge_scatter(const int* __restrict__ src, const int* __restrict__ dst,
                    const float* __restrict__ dinv, const float* __restrict__ h,
                    float* __restrict__ out, int e) {
  int eid = blockIdx.x;
  if (eid >= e) return;
  int s = src[eid], t = dst[eid];
  float nrm = dinv[s] * dinv[t];
  int d = threadIdx.x;
  atomicAdd(&out[(size_t)t * DIM + d], h[(size_t)s * DIM + d] * nrm);
}

__global__ void k_relu(float* p, int n) {
  int i = blockIdx.x * blockDim.x + threadIdx.x;
  if (i < n) p[i] = p[i] > 0.0f ? p[i] : 0.0f;
}

// ---------------------------------------------------------------------------

extern "C" void kernel_launch(void* const* d_in, const int* in_sizes, int n_in,
                              void* d_out, int out_size, void* d_ws, size_t ws_size,
                              hipStream_t stream) {
  const float* x  = (const float*)d_in[0];
  const int*   ei = (const int*)d_in[1];      // [2,E] int32 per harness convention
  const float* W1 = (const float*)d_in[2];
  const float* b1 = (const float*)d_in[3];
  const float* W2 = (const float*)d_in[4];
  const float* b2 = (const float*)d_in[5];
  const float* Wf = (const float*)d_in[6];
  const float* bf = (const float*)d_in[7];
  const int* src = ei;
  const int* dst = ei + N_EDGES;

  float* ws   = (float*)d_ws;
  float* dinv = ws;                                        // N floats
  float* bufA = ws + ((N_NODES + 255) & ~255);             // N*D (pre-agg h)
  float* bufB = bufA + (size_t)N_NODES * DIM;              // N*D (agg result)
  float* out  = (float*)d_out;

  const int ND = N_NODES * DIM;
  dim3 blk(256);
  dim3 gEl((ND + 255) / 256);
  dim3 gN((N_NODES + 255) / 256);
  dim3 gE((N_EDGES + 255) / 256);
  dim3 gGemm((N_NODES + BM - 1) / BM, DIM / BN);

  // degrees -> dinv
  k_fill1<<<gN, blk, 0, stream>>>(dinv, N_NODES);
  k_deg<<<gE, blk, 0, stream>>>(dst, dinv, N_EDGES);
  k_rsqrt<<<gN, blk, 0, stream>>>(dinv, N_NODES);

  // layer 1: h1 = x @ W1 ; agg ; +b1 ; relu
  gemm_bf16_wmma<<<gGemm, blk, 0, stream>>>(x, N_NODES, W1, DIM, bufA, DIM,
                                            N_NODES, N_NODES, nullptr, 0);
  k_self_bias<<<gEl, blk, 0, stream>>>(bufA, dinv, b1, bufB, ND);
  k_edge_scatter<<<dim3(N_EDGES), blk, 0, stream>>>(src, dst, dinv, bufA, bufB, N_EDGES);
  k_relu<<<gEl, blk, 0, stream>>>(bufB, ND);

  // layer 2: h2 = agg1 @ W2 ; agg ; +b2 ; relu
  gemm_bf16_wmma<<<gGemm, blk, 0, stream>>>(bufB, DIM, W2, DIM, bufA, DIM,
                                            N_NODES, DIM, nullptr, 0);
  k_self_bias<<<gEl, blk, 0, stream>>>(bufA, dinv, b2, bufB, ND);
  k_edge_scatter<<<dim3(N_EDGES), blk, 0, stream>>>(src, dst, dinv, bufA, bufB, N_EDGES);
  k_relu<<<gEl, blk, 0, stream>>>(bufB, ND);

  // final: out = agg2 @ Wf + bf
  gemm_bf16_wmma<<<gGemm, blk, 0, stream>>>(bufB, DIM, Wf, DIM, out, DIM,
                                            N_NODES, DIM, bf, 0);
}
